// BiLSTM_CRF_2894807958273
// MI455X (gfx1250) — matrix operand order repaired and checked by reading, not compiled, not run on hardware
//
#include <hip/hip_runtime.h>
#include <hip/hip_bf16.h>
#include <math.h>

#define VV 32000
#define EE 256
#define HH 512
#define KK 4
#define BB 32
#define TT 512

typedef __attribute__((ext_vector_type(16))) __bf16 v16bf;
typedef __attribute__((ext_vector_type(8)))  float  v8f;
typedef __attribute__((ext_vector_type(4)))  unsigned int u32x4;   // POD 128-bit

#define GM __attribute__((address_space(1)))
typedef const GM unsigned short* gwp;   // global bf16 weight pointer
typedef const GM u32x4*          gq4;   // global 128-bit load pointer

union ABf { u32x4 q[2]; v16bf v; };
union H8  { unsigned short us[8]; u32x4 q; };

__device__ __forceinline__ unsigned short f2bf(float f) {
    unsigned int u = __float_as_uint(f);
    u = (u + 0x7FFFu + ((u >> 16) & 1u)) >> 16;
    return (unsigned short)u;
}
__device__ __forceinline__ float sigm(float x) { return 1.0f / (1.0f + __expf(-x)); }

#define WMMA_BF16(Am, Bm, Cm) \
    __builtin_amdgcn_wmma_f32_16x16x32_bf16(false, (Am).v, false, (Bm).v, (short)0, (Cm), false, false)

// ---------------- weight cast (f32 -> bf16, layout preserved) ----------------
__global__ void k_cast(const float* __restrict__ src, unsigned short* __restrict__ dst, int n) {
    int i = blockIdx.x * blockDim.x + threadIdx.x;
    if (i < n) dst[i] = f2bf(src[i]);
}

// ---------------- embedding gather + cast, time-major X[T][B][E] -------------
__global__ void k_embed(const int* __restrict__ ids, const float* __restrict__ emb,
                        unsigned short* __restrict__ Xbf) {
    int idx = blockIdx.x * blockDim.x + threadIdx.x;     // one chunk of 8 elems
    int total = TT * BB * (EE / 8);
    if (idx >= total) return;
    int row = idx >> 5;                                  // (t*BB + b)
    int off = (idx & 31) * 8;
    int t = row / BB, b = row % BB;
    int tok = ids[b * TT + t];
    const float* src = emb + (size_t)tok * EE + off;
    H8 o;
    #pragma unroll
    for (int j = 0; j < 8; ++j) o.us[j] = f2bf(src[j]);
    *(u32x4*)(Xbf + (size_t)row * EE + off) = o.q;
}

// ---------------- persistent bidirectional LSTM (1 WG per direction) --------
// 512 threads = 16 waves. Wave w owns h-column tiles nt = 2w, 2w+1 (16 cols
// each); for each nt it computes BOTH batch-row tiles (mt=0,1) so every weight
// B-fragment load feeds two WMMAs. Cell state c stays in registers.
//
// Weight pointers are explicit address_space(1) so loads lower to
// global_load_b128 (LOADcnt only), and a laundered zero offset per timestep
// stops LICM from hoisting (and scratch-spilling) the invariant weight
// fragments out of the time loop. One macro expansion per nt (literal NP)
// keeps c/h array indices constant (register-resident); the memory-clobber
// asm between expansions stops the two GEMM chains from interleaving.

#define LSTM_NT(NP)                                                                 \
  {                                                                                 \
    const int hx = (wave * 2 + (NP)) * 16 + lrow;                                   \
    const v8f vz = {0.f,0.f,0.f,0.f,0.f,0.f,0.f,0.f};                               \
    v8f ai0 = vz, af0 = vz, ag0 = vz, ao0 = vz;                                     \
    v8f ai1 = vz, af1 = vz, ag1 = vz, ao1 = vz;                                     \
    const unsigned short* xr0 = xt_lds + lrow * EE;                                 \
    const unsigned short* xr1 = xt_lds + (16 + lrow) * EE;                          \
    _Pragma("unroll 1")                                                             \
    for (int k0 = 0; k0 < EE; k0 += 32) {                                           \
      const int ka = k0 + lhi * 16;                                                 \
      ABf A0, A1, B0, B1, B2, B3;                                                   \
      A0.q[0] = *(const u32x4*)(xr0 + ka); A0.q[1] = *(const u32x4*)(xr0 + ka + 8);\
      A1.q[0] = *(const u32x4*)(xr1 + ka); A1.q[1] = *(const u32x4*)(xr1 + ka + 8);\
      gwp wp = WihS + (size_t)hx * EE + (k0 + lhi * 16);                            \
      B0.q[0] = *(gq4)(wp);           B0.q[1] = *(gq4)(wp + 8);                     \
      B1.q[0] = *(gq4)(wp + gih);     B1.q[1] = *(gq4)(wp + gih + 8);               \
      B2.q[0] = *(gq4)(wp + 2*gih);   B2.q[1] = *(gq4)(wp + 2*gih + 8);             \
      B3.q[0] = *(gq4)(wp + 3*gih);   B3.q[1] = *(gq4)(wp + 3*gih + 8);             \
      ai0 = WMMA_BF16(A0, B0, ai0);  ai1 = WMMA_BF16(A1, B0, ai1);                  \
      af0 = WMMA_BF16(A0, B1, af0);  af1 = WMMA_BF16(A1, B1, af1);                  \
      ag0 = WMMA_BF16(A0, B2, ag0);  ag1 = WMMA_BF16(A1, B2, ag1);                  \
      ao0 = WMMA_BF16(A0, B3, ao0);  ao1 = WMMA_BF16(A1, B3, ao1);                  \
    }                                                                               \
    const unsigned short* hr0 = h_lds + lrow * HH;                                  \
    const unsigned short* hr1 = h_lds + (16 + lrow) * HH;                           \
    _Pragma("unroll 1")                                                             \
    for (int k0 = 0; k0 < HH; k0 += 32) {                                           \
      const int ka = k0 + lhi * 16;                                                 \
      ABf A0, A1, B0, B1, B2, B3;                                                   \
      A0.q[0] = *(const u32x4*)(hr0 + ka); A0.q[1] = *(const u32x4*)(hr0 + ka + 8);\
      A1.q[0] = *(const u32x4*)(hr1 + ka); A1.q[1] = *(const u32x4*)(hr1 + ka + 8);\
      gwp wp = WhhS + (size_t)hx * HH + (k0 + lhi * 16);                            \
      B0.q[0] = *(gq4)(wp);           B0.q[1] = *(gq4)(wp + 8);                     \
      B1.q[0] = *(gq4)(wp + ghh);     B1.q[1] = *(gq4)(wp + ghh + 8);               \
      B2.q[0] = *(gq4)(wp + 2*ghh);   B2.q[1] = *(gq4)(wp + 2*ghh + 8);             \
      B3.q[0] = *(gq4)(wp + 3*ghh);   B3.q[1] = *(gq4)(wp + 3*ghh + 8);             \
      ai0 = WMMA_BF16(A0, B0, ai0);  ai1 = WMMA_BF16(A1, B0, ai1);                  \
      af0 = WMMA_BF16(A0, B1, af0);  af1 = WMMA_BF16(A1, B1, af1);                  \
      ag0 = WMMA_BF16(A0, B2, ag0);  ag1 = WMMA_BF16(A1, B2, ag1);                  \
      ao0 = WMMA_BF16(A0, B3, ao0);  ao1 = WMMA_BF16(A1, B3, ao1);                  \
    }                                                                               \
    const float bi  = bias_lds[hx];                                                 \
    const float bfv = bias_lds[HH + hx];                                            \
    const float bg  = bias_lds[2 * HH + hx];                                        \
    const float bo  = bias_lds[3 * HH + hx];                                        \
    _Pragma("unroll")                                                               \
    for (int r = 0; r < 8; ++r) {                                                   \
      float iv = sigm(ai0[r] + bi),  fv = sigm(af0[r] + bfv);                       \
      float gv = tanhf(ag0[r] + bg), ov = sigm(ao0[r] + bo);                        \
      float cn = fv * c_reg[2*(NP)][r] + iv * gv;                                   \
      c_reg[2*(NP)][r] = cn;                                                        \
      float hv = ov * tanhf(cn);                                                    \
      hnew[2*(NP)][r] = hv;                                                         \
      hcat[(size_t)t * (BB*2*HH) + (size_t)(lhi*8 + r) * (2*HH) + dir*HH + hx] = hv;\
      iv = sigm(ai1[r] + bi);  fv = sigm(af1[r] + bfv);                             \
      gv = tanhf(ag1[r] + bg); ov = sigm(ao1[r] + bo);                              \
      cn = fv * c_reg[2*(NP)+1][r] + iv * gv;                                       \
      c_reg[2*(NP)+1][r] = cn;                                                      \
      hv = ov * tanhf(cn);                                                          \
      hnew[2*(NP)+1][r] = hv;                                                       \
      hcat[(size_t)t * (BB*2*HH) + (size_t)(16 + lhi*8 + r) * (2*HH) + dir*HH + hx] = hv; \
    }                                                                               \
  }

__launch_bounds__(512, 1)
__global__ void k_bilstm(const unsigned short* __restrict__ Xbf,
                         const unsigned short* __restrict__ Wih_f,
                         const unsigned short* __restrict__ Whh_f,
                         const float* __restrict__ b_f,
                         const unsigned short* __restrict__ Wih_b,
                         const unsigned short* __restrict__ Whh_b,
                         const float* __restrict__ b_b,
                         float* __restrict__ hcat) {
    __shared__ unsigned short xt_lds[BB * EE];     // 16 KB
    __shared__ unsigned short h_lds[BB * HH];      // 32 KB (bf16 h_{t-1})
    __shared__ float bias_lds[4 * HH];             //  8 KB

    const int dir = blockIdx.x;
    const gwp Wih = (gwp)(dir == 0 ? Wih_f : Wih_b);
    const gwp Whh = (gwp)(dir == 0 ? Whh_f : Whh_b);
    const float* bias = dir == 0 ? b_f : b_b;

    const int tid  = threadIdx.x;
    const int lane = tid & 31;
    const int wave = tid >> 5;        // 0..15
    const int lrow = lane & 15;
    const int lhi  = lane >> 4;

    for (int i = tid; i < BB * HH / 2; i += 512) ((unsigned int*)h_lds)[i] = 0u;
    for (int i = tid; i < 4 * HH; i += 512) bias_lds[i] = bias[i];

    float c_reg[4][8];
    float hnew [4][8];
    #pragma unroll
    for (int g = 0; g < 4; ++g)
        #pragma unroll
        for (int r = 0; r < 8; ++r) c_reg[g][r] = 0.0f;

    const size_t gih = (size_t)HH * EE;   // gate stride in W_ih (elements)
    const size_t ghh = (size_t)HH * HH;   // gate stride in W_hh

    __syncthreads();

    for (int s = 0; s < TT; ++s) {
        const int t = (dir == 0) ? s : (TT - 1 - s);

        { // stage x_t into LDS (1024 x 128-bit), prefetch next step
            const u32x4* src = (const u32x4*)(Xbf + (size_t)t * BB * EE);
            u32x4* dst = (u32x4*)xt_lds;
            for (int i = tid; i < BB * EE / 8; i += 512) dst[i] = src[i];
            if (s + 1 < TT) {
                const int tn = (dir == 0) ? (s + 1) : (TT - 2 - s);
                if (tid < 128)
                    __builtin_prefetch(Xbf + (size_t)tn * BB * EE + tid * 64, 0, 0);
            }
        }
        __syncthreads();

        // opaque zero offset per step: keeps the weight loads inside the time
        // loop (no LICM -> no scratch spill) while preserving addrspace(1)
        // so they lower to global_load_b128, not flat_load_b128.
        size_t lz = 0;
        asm volatile("" : "+s"(lz));
        const gwp WihS = Wih + lz;
        const gwp WhhS = Whh + lz;

        LSTM_NT(0);
        asm volatile("" ::: "memory");   // serialize the two nt chains
        LSTM_NT(1);

        __syncthreads();   // all waves done reading h_{t-1}

        #pragma unroll
        for (int np = 0; np < 2; ++np) {
            const int hx = (wave * 2 + np) * 16 + lrow;
            #pragma unroll
            for (int r = 0; r < 8; ++r) {
                h_lds[(lhi * 8 + r) * HH + hx]        = f2bf(hnew[2 * np][r]);
                h_lds[(16 + lhi * 8 + r) * HH + hx]   = f2bf(hnew[2 * np + 1][r]);
            }
        }
        __syncthreads();   // h_t visible before next step's GEMM
    }
}

// ---------------- LayerNorm + ReLU + emission (one wave per token) ----------
__global__ void k_ln_emis(const float* __restrict__ hcat,
                          const float* __restrict__ lng, const float* __restrict__ lnb,
                          const float* __restrict__ Wout, const float* __restrict__ bout,
                          float* __restrict__ emis) {
    const int wv   = (blockIdx.x * blockDim.x + threadIdx.x) >> 5;
    const int lane = threadIdx.x & 31;
    if (wv >= TT * BB) return;
    const float* row = hcat + (size_t)wv * 2 * HH;
    float s = 0.f, ss = 0.f;
    for (int j = lane; j < 2 * HH; j += 32) { float v = row[j]; s += v; ss += v * v; }
    #pragma unroll
    for (int m = 16; m; m >>= 1) { s += __shfl_xor(s, m, 32); ss += __shfl_xor(ss, m, 32); }
    const float mu  = s * (1.0f / (2 * HH));
    const float var = ss * (1.0f / (2 * HH)) - mu * mu;
    const float inv = rsqrtf(var + 1e-5f);
    float d0 = 0.f, d1 = 0.f, d2 = 0.f, d3 = 0.f;
    for (int j = lane; j < 2 * HH; j += 32) {
        float v = (row[j] - mu) * inv * lng[j] + lnb[j];
        v = v > 0.f ? v : 0.f;
        d0 += v * Wout[j];
        d1 += v * Wout[1 * 2 * HH + j];
        d2 += v * Wout[2 * 2 * HH + j];
        d3 += v * Wout[3 * 2 * HH + j];
    }
    #pragma unroll
    for (int m = 16; m; m >>= 1) {
        d0 += __shfl_xor(d0, m, 32); d1 += __shfl_xor(d1, m, 32);
        d2 += __shfl_xor(d2, m, 32); d3 += __shfl_xor(d3, m, 32);
    }
    if (lane == 0) {
        float* e = emis + (size_t)wv * KK;
        e[0] = d0 + bout[0]; e[1] = d1 + bout[1];
        e[2] = d2 + bout[2]; e[3] = d3 + bout[3];
    }
}

// ---------------- CRF NLL (one lane per batch element) -----------------------
__global__ void k_crf(const float* __restrict__ emis, const int* __restrict__ tags,
                      const int* __restrict__ mask, const float* __restrict__ st,
                      const float* __restrict__ en, const float* __restrict__ tr,
                      float* __restrict__ out) {
    __shared__ float llh[BB];
    const int b = threadIdx.x;
    if (b < BB) {
        float trm[16];
        #pragma unroll
        for (int i = 0; i < 16; ++i) trm[i] = tr[i];

        int   tag0 = tags[b * TT];
        float num  = st[tag0] + emis[(size_t)b * KK + tag0];
        int   prev = tag0;
        int   msum = (mask[b * TT] != 0) ? 1 : 0;

        float score[KK];
        #pragma unroll
        for (int j = 0; j < KK; ++j) score[j] = st[j] + emis[(size_t)b * KK + j];

        for (int t = 1; t < TT; ++t) {
            const int   tg = tags[b * TT + t];
            const int   mk = mask[b * TT + t];
            const float* e = emis + ((size_t)t * BB + b) * KK;
            if (mk) { num += trm[prev * KK + tg] + e[tg]; msum += 1; }
            prev = tg;
            float nxt[KK];
            #pragma unroll
            for (int j = 0; j < KK; ++j) {
                float m = score[0] + trm[0 * KK + j];
                #pragma unroll
                for (int i = 1; i < KK; ++i) {
                    float v = score[i] + trm[i * KK + j];
                    m = v > m ? v : m;
                }
                float sum = 0.f;
                #pragma unroll
                for (int i = 0; i < KK; ++i) sum += __expf(score[i] + trm[i * KK + j] - m);
                nxt[j] = m + __logf(sum) + e[j];
            }
            if (mk) {
                #pragma unroll
                for (int j = 0; j < KK; ++j) score[j] = nxt[j];
            }
        }
        const int ltag = tags[b * TT + (msum - 1)];
        num += en[ltag];
        float m = score[0] + en[0];
        #pragma unroll
        for (int j = 1; j < KK; ++j) { float v = score[j] + en[j]; m = v > m ? v : m; }
        float sum = 0.f;
        #pragma unroll
        for (int j = 0; j < KK; ++j) sum += __expf(score[j] + en[j] - m);
        const float den = m + __logf(sum);
        llh[b] = num - den;
    }
    __syncthreads();
    if (threadIdx.x == 0) {
        float a = 0.f;
        for (int i = 0; i < BB; ++i) a += llh[i];
        out[0] = -a / (float)BB;
    }
}

// ---------------- launcher ---------------------------------------------------
extern "C" void kernel_launch(void* const* d_in, const int* in_sizes, int n_in,
                              void* d_out, int out_size, void* d_ws, size_t ws_size,
                              hipStream_t stream) {
    (void)in_sizes; (void)n_in; (void)out_size; (void)ws_size;
    const int*   ids   = (const int*)d_in[0];
    const int*   tags  = (const int*)d_in[1];
    const int*   amask = (const int*)d_in[2];
    const float* emb   = (const float*)d_in[3];
    const float* Wih_f = (const float*)d_in[4];
    const float* Whh_f = (const float*)d_in[5];
    const float* b_f   = (const float*)d_in[6];
    const float* Wih_b = (const float*)d_in[7];
    const float* Whh_b = (const float*)d_in[8];
    const float* b_b   = (const float*)d_in[9];
    const float* ln_g  = (const float*)d_in[10];
    const float* ln_b  = (const float*)d_in[11];
    const float* Wout  = (const float*)d_in[12];
    const float* bout  = (const float*)d_in[13];
    const float* cst   = (const float*)d_in[14];
    const float* cen   = (const float*)d_in[15];
    const float* ctr   = (const float*)d_in[16];

    char* w = (char*)d_ws;
    unsigned short* Xbf     = (unsigned short*)w; w += (size_t)TT * BB * EE * 2;   //  8 MB
    unsigned short* Wihf_bf = (unsigned short*)w; w += (size_t)4 * HH * EE * 2;    //  1 MB
    unsigned short* Whhf_bf = (unsigned short*)w; w += (size_t)4 * HH * HH * 2;    //  2 MB
    unsigned short* Wihb_bf = (unsigned short*)w; w += (size_t)4 * HH * EE * 2;    //  1 MB
    unsigned short* Whhb_bf = (unsigned short*)w; w += (size_t)4 * HH * HH * 2;    //  2 MB
    float*          hcat    = (float*)w;          w += (size_t)TT * BB * 2 * HH * 4; // 64 MB
    float*          emis    = (float*)w;          w += (size_t)TT * BB * KK * 4;   // 256 KB

    const int nih = 4 * HH * EE, nhh = 4 * HH * HH;
    k_cast<<<(nih + 255) / 256, 256, 0, stream>>>(Wih_f, Wihf_bf, nih);
    k_cast<<<(nhh + 255) / 256, 256, 0, stream>>>(Whh_f, Whhf_bf, nhh);
    k_cast<<<(nih + 255) / 256, 256, 0, stream>>>(Wih_b, Wihb_bf, nih);
    k_cast<<<(nhh + 255) / 256, 256, 0, stream>>>(Whh_b, Whhb_bf, nhh);

    const int nchunks = TT * BB * (EE / 8);
    k_embed<<<(nchunks + 255) / 256, 256, 0, stream>>>(ids, emb, Xbf);

    k_bilstm<<<2, 512, 0, stream>>>(Xbf, Wihf_bf, Whhf_bf, b_f,
                                    Wihb_bf, Whhb_bf, b_b, hcat);

    k_ln_emis<<<(TT * BB) / 8, 256, 0, stream>>>(hcat, ln_g, ln_b, Wout, bout, emis);

    k_crf<<<1, 32, 0, stream>>>(emis, tags, amask, cst, cen, ctr, (float*)d_out);
}
